// VSSBlock_16939351015745
// MI455X (gfx1250) — matrix operand order repaired and checked by reading, not compiled, not run on hardware
//
#include <hip/hip_runtime.h>
#include <math.h>

typedef __attribute__((ext_vector_type(16))) __bf16 v16bf;
typedef __attribute__((ext_vector_type(8)))  float  v8f;

// ---------------------------------------------------------------------------
// WMMA GEMM: out[m,n] = (resid ? resid[m,n] : 0) + sum_k A[m,k] * W[n,k]
// A: [M,K] fp32 (lda), W: [N,K] fp32 (ldw) -> converted to bf16 on the fly.
// One wave computes a 16x64 tile (4 accumulators): the A fragment is loaded
// once per K-step and reused by 4 WMMA issues. K must be a multiple of 32.
// N-edge handling: B row pointers are clamped to row N-1 (K-invariant), so the
// inner loop is branch-free; out-of-range columns are masked at the store.
// ---------------------------------------------------------------------------
__device__ __forceinline__ v16bf load_bfrag(const float* __restrict__ wp)
{
    v16bf b;
    #pragma unroll
    for (int e = 0; e < 16; ++e) b[e] = (__bf16)wp[e];
    return b;
}

__global__ void wmma_gemm_kernel(const float* __restrict__ A,
                                 const float* __restrict__ W,
                                 const float* __restrict__ resid,
                                 float* __restrict__ out,
                                 int M, int N, int K,
                                 int lda, int ldw, int ldr, int ldc)
{
    const int lane = threadIdx.x & 31;
    const int wave = threadIdx.x >> 5;
    const int mtiles  = (M + 15) >> 4;
    const int ntiles4 = (N + 63) >> 6;
    const long tile = (long)blockIdx.x * (blockDim.x >> 5) + wave;
    if (tile >= (long)mtiles * ntiles4) return;
    const int mt  = (int)(tile / ntiles4);   // nt-fastest: waves in a block share A rows
    const int nt4 = (int)(tile % ntiles4);

    const int hi    = lane >> 4;         // half-wave select
    const int l15   = lane & 15;
    const int mrow  = mt * 16 + l15;     // A fragment row (both halves)
    const int nbase = nt4 * 64 + l15;    // first B fragment row of W

    // K-invariant clamped B row pointers (garbage columns masked at store)
    const int n0 = (nbase            < N) ? nbase            : N - 1;
    const int n1 = (nbase + 16       < N) ? nbase + 16       : N - 1;
    const int n2 = (nbase + 32       < N) ? nbase + 32       : N - 1;
    const int n3 = (nbase + 48       < N) ? nbase + 48       : N - 1;
    const float* wp0 = W + (size_t)n0 * ldw + hi * 16;
    const float* wp1 = W + (size_t)n1 * ldw + hi * 16;
    const float* wp2 = W + (size_t)n2 * ldw + hi * 16;
    const float* wp3 = W + (size_t)n3 * ldw + hi * 16;
    const float* ap  = A + (size_t)mrow * lda + hi * 16;

    v8f acc0 = {}, acc1 = {}, acc2 = {}, acc3 = {};
    for (int k0 = 0; k0 < K; k0 += 32) {
        // A fragment: elems 0..15 -> K = k0 + hi*16 + e
        v16bf a;
        #pragma unroll
        for (int e = 0; e < 16; ++e) {
            a[e]     = (__bf16)ap[k0 + e];
        }
        // B fragments: lane = n, elems 0..15 -> K = k0 + hi*16 + e
        v16bf b0 = load_bfrag(wp0 + k0);
        v16bf b1 = load_bfrag(wp1 + k0);
        v16bf b2 = load_bfrag(wp2 + k0);
        v16bf b3 = load_bfrag(wp3 + k0);
        acc0 = __builtin_amdgcn_wmma_f32_16x16x32_bf16(false, a, false, b0, (short)0, acc0, false, false);
        acc1 = __builtin_amdgcn_wmma_f32_16x16x32_bf16(false, a, false, b1, (short)0, acc1, false, false);
        acc2 = __builtin_amdgcn_wmma_f32_16x16x32_bf16(false, a, false, b2, (short)0, acc2, false, false);
        acc3 = __builtin_amdgcn_wmma_f32_16x16x32_bf16(false, a, false, b3, (short)0, acc3, false, false);
    }

    // C/D layout: vgpr r -> row = mt*16 + r + 8*hi ; col = tile_base + (lane&15)
    const bool has_res = (resid != nullptr);
    v8f accs[4] = {acc0, acc1, acc2, acc3};
    #pragma unroll
    for (int s = 0; s < 4; ++s) {
        const int col = nt4 * 64 + s * 16 + l15;
        if (col < N) {
            #pragma unroll
            for (int r = 0; r < 8; ++r) {
                const int row = mt * 16 + r + 8 * hi;
                float v = accs[s][r];
                if (has_res) v += resid[(size_t)row * ldr + col];
                out[(size_t)row * ldc + col] = v;
            }
        }
    }
}

// ---------------------------------------------------------------------------
// LayerNorm over last dim C (C multiple of 32), one wave per row.
// ---------------------------------------------------------------------------
__global__ void ln_kernel(const float* __restrict__ x, const float* __restrict__ g,
                          const float* __restrict__ bta, float* __restrict__ out,
                          int rows, int C)
{
    int w = (blockIdx.x * blockDim.x + threadIdx.x) >> 5;
    int lane = threadIdx.x & 31;
    if (w >= rows) return;
    const float* xr = x + (size_t)w * C;
    int per = C >> 5;
    float s = 0.f, ss = 0.f;
    for (int i = 0; i < per; ++i) { float v = xr[lane + 32 * i]; s += v; ss += v * v; }
    for (int off = 16; off; off >>= 1) { s += __shfl_xor(s, off, 32); ss += __shfl_xor(ss, off, 32); }
    float mean = s / C;
    float inv  = rsqrtf(ss / C - mean * mean + 1e-5f);
    float* orow = out + (size_t)w * C;
    for (int i = 0; i < per; ++i) {
        int c = lane + 32 * i;
        orow[c] = (xr[c] - mean) * inv * g[c] + bta[c];
    }
}

// ---------------------------------------------------------------------------
// Depthwise 3x3 conv + bias + SiLU on first Di channels of xz ([B*L, ld_in]).
// ---------------------------------------------------------------------------
__global__ void dwconv2d_silu_kernel(const float* __restrict__ xz, const float* __restrict__ w,
                                     const float* __restrict__ bias, float* __restrict__ xs,
                                     int Bb, int Hh, int Ww, int Di, int ld_in)
{
    int idx = blockIdx.x * blockDim.x + threadIdx.x;
    int total = Bb * Hh * Ww * Di;
    if (idx >= total) return;
    int d = idx % Di; int p = idx / Di;
    int wx = p % Ww; int rest = p / Ww; int hy = rest % Hh; int b = rest / Hh;
    float acc = bias[d];
    #pragma unroll
    for (int dy = -1; dy <= 1; ++dy) {
        int hh = hy + dy; if (hh < 0 || hh >= Hh) continue;
        #pragma unroll
        for (int dx = -1; dx <= 1; ++dx) {
            int ww2 = wx + dx; if (ww2 < 0 || ww2 >= Ww) continue;
            acc += xz[((size_t)((b * Hh + hh) * Ww + ww2)) * ld_in + d]
                 * w[d * 9 + (dy + 1) * 3 + (dx + 1)];
        }
    }
    xs[idx] = acc / (1.f + expf(-acc));   // SiLU
}

// ---------------------------------------------------------------------------
// SimpleGate1D on channels [12..27] (w1b/w2b) and [28..43] (w1c/w2c) of dbl
// (in-place). One block of 96 threads per position.
// ---------------------------------------------------------------------------
__global__ void sgate_kernel(float* __restrict__ dbl,
                             const float* __restrict__ w1b, const float* __restrict__ w2b,
                             const float* __restrict__ w1c, const float* __restrict__ w2c,
                             int rows)
{
    __shared__ float xin[16];
    __shared__ float gh[96];
    int pos = blockIdx.x;
    if (pos >= rows) return;
    int t = threadIdx.x;   // 0..95
    float* row = dbl + (size_t)pos * 44;
    for (int gate = 0; gate < 2; ++gate) {
        const float* w1 = gate ? w1c : w1b;
        const float* w2 = gate ? w2c : w2b;
        int base = gate ? 28 : 12;
        if (t < 16) xin[t] = row[base + t];
        __syncthreads();
        float y1 = 0.f, y2 = 0.f;
        #pragma unroll
        for (int k = 0; k < 16; ++k) {
            float xv = xin[k];
            y1 += w1[t * 16 + k] * xv;
            y2 += w1[(96 + t) * 16 + k] * xv;
        }
        float gl = 0.5f * y1 * (1.f + erff(y1 * 0.70710678f));  // exact GELU
        gh[t] = gl * y2;
        __syncthreads();
        if (t < 16) {
            float o = 0.f;
            #pragma unroll
            for (int h = 0; h < 96; ++h) o += w2[t * 96 + h] * gh[h];
            row[base + t] = o;
        }
        __syncthreads();
    }
}

// ---------------------------------------------------------------------------
// Residual dilated (dil=2) depthwise conv1d along L for the 44 dt/B/C channels.
// ---------------------------------------------------------------------------
__global__ void dwconv1d_kernel(const float* __restrict__ in, float* __restrict__ out,
                                const float* __restrict__ wdt, const float* __restrict__ wB,
                                const float* __restrict__ wC, int Bb, int L)
{
    int idx = blockIdx.x * blockDim.x + threadIdx.x;
    int total = Bb * L * 44;
    if (idx >= total) return;
    int c = idx % 44; int p = idx / 44; int l = p % L; int b = p / L;
    const float* wv;
    if (c < 12)      wv = wdt + c * 15;
    else if (c < 28) wv = wB + (c - 12) * 15;
    else             wv = wC + (c - 28) * 15;
    float acc = in[idx];                  // residual
    #pragma unroll
    for (int j = 0; j < 15; ++j) {
        int ll = l + 2 * (j - 7);
        if (ll >= 0 && ll < L)
            acc += in[((size_t)(b * L + ll)) * 44 + c] * wv[j];
    }
    out[idx] = acc;
}

// ---------------------------------------------------------------------------
// delta = softplus(dbc[:, :12] @ dt_projs_w.T + dt_projs_b)
// ---------------------------------------------------------------------------
__global__ void delta_kernel(const float* __restrict__ dbc, const float* __restrict__ dtw,
                             const float* __restrict__ dtb, float* __restrict__ delta,
                             int rows, int Di)
{
    int idx = blockIdx.x * blockDim.x + threadIdx.x;
    if (idx >= rows * Di) return;
    int d = idx % Di; int p = idx / Di;
    const float* r = dbc + (size_t)p * 44;
    float a = dtb[d];
    #pragma unroll
    for (int k = 0; k < 12; ++k) a += dtw[d * 12 + k] * r[k];
    delta[idx] = (a > 20.f) ? a : log1pf(expf(a));
}

// ---------------------------------------------------------------------------
// Selective scan (S6): one thread per (b,d) channel, 16 states in registers.
// Sequential over L; exp(dt*A) via exp2f with A pre-scaled by log2(e).
// ---------------------------------------------------------------------------
__global__ void scan_kernel(const float* __restrict__ u, const float* __restrict__ delta,
                            const float* __restrict__ dbc, const float* __restrict__ A_logs,
                            const float* __restrict__ Ds, float* __restrict__ y,
                            int L, int Di)
{
    int d = blockIdx.x * blockDim.x + threadIdx.x;
    int b = blockIdx.y;
    if (d >= Di) return;
    float a2[16], h[16];
    #pragma unroll
    for (int n = 0; n < 16; ++n) {
        a2[n] = -expf(A_logs[d * 16 + n]) * 1.44269504f;   // A * log2(e)
        h[n] = 0.f;
    }
    const float Dd = Ds[d];
    const size_t rowB = (size_t)b * L;
    for (int t = 0; t < L; ++t) {
        size_t pr = rowB + t;
        if (t + 8 < L)   // hide HBM latency on the serial chain
            __builtin_prefetch(dbc + (pr + 8) * 44, 0, 1);
        float dt = delta[pr * Di + d];
        float uv = u[pr * Di + d];
        const float* bc = dbc + pr * 44;
        float du = dt * uv;
        float acc = 0.f;
        #pragma unroll
        for (int n = 0; n < 16; ++n) {
            h[n] = exp2f(dt * a2[n]) * h[n] + du * bc[12 + n];
            acc += h[n] * bc[28 + n];
        }
        y[pr * Di + d] = acc + uv * Dd;
    }
}

// ---------------------------------------------------------------------------
// out-norm (LayerNorm over Di) * SiLU(z), one wave per position.
// ---------------------------------------------------------------------------
__global__ void normgate_kernel(const float* __restrict__ y, const float* __restrict__ xz,
                                const float* __restrict__ g, const float* __restrict__ bta,
                                float* __restrict__ yg, int rows, int Di, int ld_xz)
{
    int w = (blockIdx.x * blockDim.x + threadIdx.x) >> 5;
    int lane = threadIdx.x & 31;
    if (w >= rows) return;
    const float* yr = y + (size_t)w * Di;
    int per = Di >> 5;
    float s = 0.f, ss = 0.f;
    for (int i = 0; i < per; ++i) { float v = yr[lane + 32 * i]; s += v; ss += v * v; }
    for (int off = 16; off; off >>= 1) { s += __shfl_xor(s, off, 32); ss += __shfl_xor(ss, off, 32); }
    float mean = s / Di;
    float inv  = rsqrtf(ss / Di - mean * mean + 1e-5f);
    const float* zr = xz + (size_t)w * ld_xz + Di;   // z = second half of xz
    float* o = yg + (size_t)w * Di;
    for (int i = 0; i < per; ++i) {
        int c = lane + 32 * i;
        float zv = zr[c];
        float sz = zv / (1.f + expf(-zv));
        o[c] = ((yr[c] - mean) * inv * g[c] + bta[c]) * sz;
    }
}

// ---------------------------------------------------------------------------

static inline void launch_gemm(const float* A, const float* W, const float* resid,
                               float* out, int M, int N, int K,
                               int lda, int ldw, int ldr, int ldc, hipStream_t s)
{
    int mt = (M + 15) / 16, nt4 = (N + 63) / 64;
    long tiles = (long)mt * nt4;
    int blocks = (int)((tiles + 3) / 4);     // 4 waves / 128-thread block
    hipLaunchKernelGGL(wmma_gemm_kernel, dim3(blocks), dim3(128), 0, s,
                       A, W, resid, out, M, N, K, lda, ldw, ldr, ldc);
}

extern "C" void kernel_launch(void* const* d_in, const int* in_sizes, int n_in,
                              void* d_out, int out_size, void* d_ws, size_t ws_size,
                              hipStream_t stream)
{
    (void)in_sizes; (void)n_in; (void)out_size; (void)ws_size;

    const float* x             = (const float*)d_in[0];
    const float* low           = (const float*)d_in[1];
    const float* ln_g          = (const float*)d_in[2];
    const float* ln_b          = (const float*)d_in[3];
    const float* in_proj_w     = (const float*)d_in[4];
    const float* in_proj_low_w = (const float*)d_in[5];
    const float* conv2d_w      = (const float*)d_in[6];
    const float* conv2d_b      = (const float*)d_in[7];
    const float* x_proj_w      = (const float*)d_in[8];
    const float* x_proj_w_low  = (const float*)d_in[9];
    const float* conv_dt_w     = (const float*)d_in[10];
    const float* conv_B_w      = (const float*)d_in[11];
    const float* conv_C_w      = (const float*)d_in[12];
    const float* sgb_w1        = (const float*)d_in[13];
    const float* sgb_w2        = (const float*)d_in[14];
    const float* sgc_w1        = (const float*)d_in[15];
    const float* sgc_w2        = (const float*)d_in[16];
    const float* dt_projs_w    = (const float*)d_in[17];
    const float* dt_projs_b    = (const float*)d_in[18];
    const float* A_logs        = (const float*)d_in[19];
    const float* Ds            = (const float*)d_in[20];
    const float* out_norm_g    = (const float*)d_in[21];
    const float* out_norm_b    = (const float*)d_in[22];
    const float* out_proj_w    = (const float*)d_in[23];
    float* out = (float*)d_out;

    const int Bb = 4, Hh = 64, Ww = 64, Cm = 192, Di = 384;
    const int L = Hh * Ww;            // 4096
    const int rows = Bb * L;          // 16384
    const int DBL = 44;               // DT_RANK + 2*D_STATE

    // workspace layout (fp32), with temporal reuse: xn<->yg, delta<->lowp
    float* ws   = (float*)d_ws;
    float* xz   = ws;                              // rows * 768
    float* xs   = xz   + (size_t)rows * 768;       // rows * 384
    float* lowp = xs   + (size_t)rows * 384;       // rows * 384 (later: delta)
    float* yb   = lowp + (size_t)rows * 384;       // rows * 384
    float* yg   = yb   + (size_t)rows * 384;       // rows * 384 (early: xn)
    float* dblL = yg   + (size_t)rows * 384;       // rows * 44
    float* dblS = dblL + (size_t)rows * DBL;       // rows * 44
    float* dbc  = dblS + (size_t)rows * DBL;       // rows * 44
    float* xn    = yg;     // dead before yg written
    float* delta = lowp;   // lowp dead before delta written

    // 1) pre-LN of x
    hipLaunchKernelGGL(ln_kernel, dim3(rows / 8), dim3(256), 0, stream,
                       x, ln_g, ln_b, xn, rows, Cm);
    // 2) xz = xn @ in_proj_w.T           [16384 x 768], K=192  (WMMA)
    launch_gemm(xn, in_proj_w, nullptr, xz, rows, 2 * Di, Cm, Cm, Cm, 0, 2 * Di, stream);
    // 3) lowp = low @ in_proj_low_w.T    [16384 x 384], K=192  (WMMA)
    launch_gemm(low, in_proj_low_w, nullptr, lowp, rows, Di, Cm, Cm, Cm, 0, Di, stream);
    // 4) xs = SiLU(dwconv3x3(xh) + b)
    {
        int total = rows * Di;
        hipLaunchKernelGGL(dwconv2d_silu_kernel, dim3((total + 255) / 256), dim3(256), 0, stream,
                           xz, conv2d_w, conv2d_b, xs, Bb, Hh, Ww, Di, 2 * Di);
    }
    // 5) low_dbl = lowp @ x_proj_w_low.T  [16384 x 44], K=384  (WMMA, padded N)
    launch_gemm(lowp, x_proj_w_low, nullptr, dblL, rows, DBL, Di, Di, Di, 0, DBL, stream);
    // 6) SimpleGate on B/C channels of low_dbl (in place)
    hipLaunchKernelGGL(sgate_kernel, dim3(rows), dim3(96), 0, stream,
                       dblL, sgb_w1, sgb_w2, sgc_w1, sgc_w2, rows);
    // 7) dblS = low_dbl + xs @ x_proj_w.T  (residual fused in GEMM epilogue)
    launch_gemm(xs, x_proj_w, dblL, dblS, rows, DBL, Di, Di, Di, DBL, DBL, stream);
    // 8) residual dilated depthwise conv1d along L on dt/B/C
    {
        int total = rows * DBL;
        hipLaunchKernelGGL(dwconv1d_kernel, dim3((total + 255) / 256), dim3(256), 0, stream,
                           dblS, dbc, conv_dt_w, conv_B_w, conv_C_w, Bb, L);
    }
    // 9) delta = softplus(dt @ dt_projs_w.T + dt_projs_b)
    {
        int total = rows * Di;
        hipLaunchKernelGGL(delta_kernel, dim3((total + 255) / 256), dim3(256), 0, stream,
                           dbc, dt_projs_w, dt_projs_b, delta, rows, Di);
    }
    // 10) selective scan (sequential over L)
    hipLaunchKernelGGL(scan_kernel, dim3(Di / 128, Bb), dim3(128), 0, stream,
                       xs, delta, dbc, A_logs, Ds, yb, L, Di);
    // 11) yg = LN(y) * SiLU(z)
    hipLaunchKernelGGL(normgate_kernel, dim3(rows / 8), dim3(256), 0, stream,
                       yb, xz, out_norm_g, out_norm_b, yg, rows, Di, 2 * Di);
    // 12) out = x + yg @ out_proj_w.T    [16384 x 192], K=384  (WMMA, fused residual)
    launch_gemm(yg, out_proj_w, x, out, rows, Cm, Di, Di, Di, Cm, Cm, stream);
}
